// ChangeDecoder_79542794322019
// MI455X (gfx1250) — matrix operand order repaired and checked by reading, not compiled, not run on hardware
//
#include <hip/hip_runtime.h>
#include <hip/hip_bf16.h>
#include <stdint.h>

// ---------------------------------------------------------------------------
// CDNA5 (gfx1250) implementation of the ChangeDecoder reference.
// All GEMM-like ops (1x1 convs, in/out/x/dt projections, im2col'd 3x3 convs,
// SE FCs) run through a single wave32 WMMA kernel: v_wmma_f32_16x16x32_f16
// with f32 accumulation, fused bias/BN/residual/activation epilogue.
// Each wave computes a 16x32 output tile (2 WMMA accumulators, shared A frag).
// Hot loop is branch-free: full K-tiles use unmasked vectorized staging; a
// masked tail tile handles ragged K (only the tiny K=8/K=16 GEMMs hit it).
// ---------------------------------------------------------------------------

typedef __attribute__((ext_vector_type(16))) _Float16     v16h;
typedef __attribute__((ext_vector_type(8)))  float        v8f;
typedef __attribute__((ext_vector_type(4)))  unsigned int u32x4;

#define ACT_NONE     0
#define ACT_RELU     1
#define ACT_SILU     2
#define ACT_SIGMOID  3
#define ACT_SOFTPLUS 4

__device__ __forceinline__ float act_apply(float v, int act) {
  switch (act) {
    case ACT_RELU:     return v > 0.f ? v : 0.f;
    case ACT_SILU:     return v / (1.f + __expf(-v));
    case ACT_SIGMOID:  return 1.f / (1.f + __expf(-v));
    case ACT_SOFTPLUS: return (v > 20.f) ? v : log1pf(__expf(v));
    default:           return v;
  }
}

__device__ __forceinline__ uint32_t pk_f16x2(float a, float b) {
  union { _Float16 h[2]; uint32_t u; } x;
  x.h[0] = (_Float16)a; x.h[1] = (_Float16)b;
  return x.u;
}

// Gather A/B fragments from per-wave LDS tiles and issue the two WMMAs.
// A 16x32 f16 frag: pair p -> kpair = half*8 + p                    (2x b128)
// B 32x16 f16 frag: pair p -> kpair = half*8 + p                    (2x b128 each)
__device__ __forceinline__ void wmma_step(const uint32_t* la, const uint32_t* lb,
                                          int mn, int half, v8f& acc0, v8f& acc1) {
  union { v16h v; u32x4 q[2]; } af, bf0, bf1;
  af.q[0]  = ((const u32x4*)la)[mn * 4 + half * 2];
  af.q[1]  = ((const u32x4*)la)[mn * 4 + half * 2 + 1];
  bf0.q[0] = ((const u32x4*)lb)[mn * 4 + half * 2];
  bf0.q[1] = ((const u32x4*)lb)[mn * 4 + half * 2 + 1];
  bf1.q[0] = ((const u32x4*)lb)[(16 + mn) * 4 + half * 2];
  bf1.q[1] = ((const u32x4*)lb)[(16 + mn) * 4 + half * 2 + 1];
  acc0 = __builtin_amdgcn_wmma_f32_16x16x32_f16(false, af.v, false, bf0.v,
                                                (short)0, acc0, false, false);
  acc1 = __builtin_amdgcn_wmma_f32_16x16x32_f16(false, af.v, false, bf1.v,
                                                (short)0, acc1, false, false);
}

// ---------------------------------------------------------------------------
// WMMA GEMM:  C[M,N] = act( (A[M,K] @ B + bias) * bnScale + bnShift + res )
//   A: f32 row-major, row stride lda (lda % 4 == 0 at every call site)
//   B: f32, bT=1 -> stored [N,K] (OIHW conv weights), bT=0 -> [K,N]
// 4 waves / block; each wave owns a 16(M) x 32(N) tile.
// OOB M rows / N cols are clamped to row/col 0 (their outputs are never
// stored), so only the K dimension ever needs value masking (tail tile).
// ---------------------------------------------------------------------------
__global__ __launch_bounds__(128)
void k_gemm(const float* __restrict__ A, int lda,
            const float* __restrict__ Bm, int bT,
            const float* __restrict__ bias,
            const float* __restrict__ res,
            const float* __restrict__ bng, const float* __restrict__ bnb,
            const float* __restrict__ bnm, const float* __restrict__ bnv,
            float* __restrict__ Cout, int M, int N, int K, int act)
{
  // per-wave LDS: A tile 16x32 f16 = 256 u32; B tile 32(N)x32(K) f16 = 512 u32
  __shared__ __align__(16) uint32_t lA[4][256];
  __shared__ __align__(16) uint32_t lB[4][512];

  const int wave = threadIdx.x >> 5;
  const int lane = threadIdx.x & 31;
  const int tilesN2 = (N + 31) >> 5;
  const int tilesM  = (M + 15) >> 4;
  const int nT = tilesM * tilesN2;
  int t = blockIdx.x * 4 + wave;
  const bool active = (t < nT);
  if (!active) t = 0;
  const int m0  = (t / tilesN2) << 4;
  const int n0w = (t % tilesN2) << 5;

  const int half = lane >> 4;
  const int mn   = lane & 15;
  uint32_t* la = lA[wave];
  uint32_t* lb = lB[wave];

  v8f acc0 = {};
  v8f acc1 = {};

  // A-tile mapping: lane covers row = lane/2, 16 consecutive k values
  const int a_row  = lane >> 1;
  const int a_koff = (lane & 1) << 4;
  const int mrow   = m0 + a_row;
  const size_t a_base = (size_t)(mrow < M ? mrow : 0) * lda;
  // B-tile mapping: lane covers n-row = lane, 32 consecutive k values
  const int  ncol = n0w + lane;
  const size_t b_base = bT ? (size_t)(ncol < N ? ncol : 0) * K
                           : (size_t)(ncol < N ? ncol : 0);

  const int Kfull = K & ~31;
  const bool bvec = bT && ((K & 3) == 0);   // float4-aligned weight rows

  for (int k0 = 0; k0 < Kfull; k0 += 32) {
    // ---- stage A: 4x global_load_b128, pack, 2x ds_store_b128 ----
    {
      const float4* ap = (const float4*)(A + a_base + k0 + a_koff);
      const float4 f0 = ap[0], f1 = ap[1], f2 = ap[2], f3 = ap[3];
      u32x4 v0 = { pk_f16x2(f0.x, f0.y), pk_f16x2(f0.z, f0.w),
                   pk_f16x2(f1.x, f1.y), pk_f16x2(f1.z, f1.w) };
      u32x4 v1 = { pk_f16x2(f2.x, f2.y), pk_f16x2(f2.z, f2.w),
                   pk_f16x2(f3.x, f3.y), pk_f16x2(f3.z, f3.w) };
      ((u32x4*)la)[lane * 2 + 0] = v0;
      ((u32x4*)la)[lane * 2 + 1] = v1;
    }
    // ---- stage B: vectorized for [N,K] weights, strided scalar for [K,N] ----
    if (bvec) {
      const float4* bp = (const float4*)(Bm + b_base + k0);
#pragma unroll
      for (int q4 = 0; q4 < 4; ++q4) {
        const float4 g0 = bp[2 * q4], g1 = bp[2 * q4 + 1];
        u32x4 v = { pk_f16x2(g0.x, g0.y), pk_f16x2(g0.z, g0.w),
                    pk_f16x2(g1.x, g1.y), pk_f16x2(g1.z, g1.w) };
        ((u32x4*)lb)[lane * 4 + q4] = v;
      }
    } else if (bT) {
#pragma unroll
      for (int q4 = 0; q4 < 4; ++q4) {
        const float* bp = Bm + b_base + k0 + 8 * q4;
        u32x4 v = { pk_f16x2(bp[0], bp[1]), pk_f16x2(bp[2], bp[3]),
                    pk_f16x2(bp[4], bp[5]), pk_f16x2(bp[6], bp[7]) };
        ((u32x4*)lb)[lane * 4 + q4] = v;
      }
    } else {
#pragma unroll
      for (int q4 = 0; q4 < 4; ++q4) {
        const float* bp = Bm + (size_t)(k0 + 8 * q4) * N + b_base;
        const size_t sn = (size_t)N;
        u32x4 v = { pk_f16x2(bp[0],      bp[sn]),
                    pk_f16x2(bp[2 * sn], bp[3 * sn]),
                    pk_f16x2(bp[4 * sn], bp[5 * sn]),
                    pk_f16x2(bp[6 * sn], bp[7 * sn]) };
        ((u32x4*)lb)[lane * 4 + q4] = v;
      }
    }
    // prefetch next A K-tile (global_prefetch)
    if (k0 + 32 < K) __builtin_prefetch(&A[a_base + k0 + 32 + a_koff], 0, 1);

    __syncthreads();
    wmma_step(la, lb, mn, half, acc0, acc1);
    __syncthreads();
  }

  // ---- masked tail tile (K % 32 != 0; runs at most once) ----
  if (Kfull < K) {
    const int k0 = Kfull;
    {
      uint32_t tmp[8];
#pragma unroll
      for (int q = 0; q < 8; ++q) {
        const int k1 = k0 + a_koff + 2 * q;
        const int k2 = k1 + 1;
        float a0 = A[a_base + (k1 < K ? k1 : 0)];
        float a1 = A[a_base + (k2 < K ? k2 : 0)];
        a0 = (k1 < K) ? a0 : 0.f;
        a1 = (k2 < K) ? a1 : 0.f;
        tmp[q] = pk_f16x2(a0, a1);
      }
      u32x4 v0 = {tmp[0], tmp[1], tmp[2], tmp[3]};
      u32x4 v1 = {tmp[4], tmp[5], tmp[6], tmp[7]};
      ((u32x4*)la)[lane * 2 + 0] = v0;
      ((u32x4*)la)[lane * 2 + 1] = v1;
    }
    {
      uint32_t tmp[16];
#pragma unroll
      for (int q = 0; q < 16; ++q) {
        const int k1 = k0 + 2 * q;
        const int k2 = k1 + 1;
        float b0, b1;
        if (bT) {
          b0 = Bm[b_base + (k1 < K ? k1 : 0)];
          b1 = Bm[b_base + (k2 < K ? k2 : 0)];
        } else {
          b0 = Bm[(size_t)(k1 < K ? k1 : 0) * N + b_base];
          b1 = Bm[(size_t)(k2 < K ? k2 : 0) * N + b_base];
        }
        b0 = (k1 < K) ? b0 : 0.f;
        b1 = (k2 < K) ? b1 : 0.f;
        tmp[q] = pk_f16x2(b0, b1);
      }
#pragma unroll
      for (int q4 = 0; q4 < 4; ++q4) {
        u32x4 v = {tmp[4 * q4], tmp[4 * q4 + 1], tmp[4 * q4 + 2], tmp[4 * q4 + 3]};
        ((u32x4*)lb)[lane * 4 + q4] = v;
      }
    }
    __syncthreads();
    wmma_step(la, lb, mn, half, acc0, acc1);
    __syncthreads();
  }

  if (!active) return;
#pragma unroll
  for (int sub = 0; sub < 2; ++sub) {
    const int n = n0w + sub * 16 + mn;
    if (n >= N) continue;
    const v8f a = sub ? acc1 : acc0;
    float bscale = 1.f, bshift = 0.f;
    if (bng) { const float sc = bng[n] * rsqrtf(bnv[n] + 1e-5f); bscale = sc; bshift = bnb[n] - bnm[n] * sc; }
    const float bv = bias ? bias[n] : 0.f;
#pragma unroll
    for (int r = 0; r < 8; ++r) {
      const int m = m0 + half * 8 + r;   // C/D layout: lanes0-15 M=r, lanes16-31 M=8+r
      if (m >= M) continue;
      float v = a[r] + bv;
      v = v * bscale + bshift;
      if (res) v += res[(size_t)m * N + n];
      Cout[(size_t)m * N + n] = act_apply(v, act);
    }
  }
}

// ---------------------------------------------------------------------------
// Elementwise / layout kernels
// ---------------------------------------------------------------------------
__global__ void k_cat_c_nhwc(const float* __restrict__ pre, const float* __restrict__ post,
                             float* __restrict__ out, int B, int C, int H, int W) {
  int idx = blockIdx.x * blockDim.x + threadIdx.x;
  const int total = B * H * W * 2 * C;
  if (idx >= total) return;
  int c2 = idx % (2 * C); int t = idx / (2 * C);
  int w = t % W; t /= W; int h = t % H; int b = t / H;
  float v = (c2 < C) ? pre[((size_t)(b * C + c2) * H + h) * W + w]
                     : post[((size_t)(b * C + (c2 - C)) * H + h) * W + w];
  out[idx] = v;
}

// stack along last axis then reshape -> interleaved columns NHWC [B,H,2W,C]
__global__ void k_stackW_nhwc(const float* __restrict__ pre, const float* __restrict__ post,
                              float* __restrict__ out, int B, int C, int H, int W) {
  int idx = blockIdx.x * blockDim.x + threadIdx.x;
  const int total = B * H * 2 * W * C;
  if (idx >= total) return;
  int c = idx % C; int t = idx / C;
  int w2 = t % (2 * W); t /= (2 * W); int h = t % H; int b = t / H;
  int w = w2 >> 1, s = w2 & 1;
  const float* src = s ? post : pre;
  out[idx] = src[((size_t)(b * C + c) * H + h) * W + w];
}

// concat along W NHWC [B,H,2W,C]
__global__ void k_catW_nhwc(const float* __restrict__ pre, const float* __restrict__ post,
                            float* __restrict__ out, int B, int C, int H, int W) {
  int idx = blockIdx.x * blockDim.x + threadIdx.x;
  const int total = B * H * 2 * W * C;
  if (idx >= total) return;
  int c = idx % C; int t = idx / C;
  int w2 = t % (2 * W); t /= (2 * W); int h = t % H; int b = t / H;
  const float* src = (w2 < W) ? pre : post;
  int w = (w2 < W) ? w2 : w2 - W;
  out[idx] = src[((size_t)(b * C + c) * H + h) * W + w];
}

__global__ void k_cat2(const float* __restrict__ a, const float* __restrict__ b,
                       float* __restrict__ out, int M) {
  int idx = blockIdx.x * blockDim.x + threadIdx.x;
  if (idx >= M * 256) return;
  int c = idx % 256, m = idx / 256;
  out[idx] = (c < 128) ? a[m * 128 + c] : b[m * 128 + (c - 128)];
}

__global__ void k_add(const float* __restrict__ a, const float* __restrict__ b,
                      float* __restrict__ out, int n) {
  int idx = blockIdx.x * blockDim.x + threadIdx.x;
  if (idx < n) out[idx] = a[idx] + b[idx];
}

__global__ void k_mul_silu(const float* __restrict__ a, const float* __restrict__ z,
                           float* __restrict__ out, int n) {
  int idx = blockIdx.x * blockDim.x + threadIdx.x;
  if (idx < n) { float zv = z[idx]; out[idx] = a[idx] * (zv / (1.f + __expf(-zv))); }
}

__global__ void k_split(const float* __restrict__ xz, float* __restrict__ xp,
                        float* __restrict__ z, int M) {
  int idx = blockIdx.x * blockDim.x + threadIdx.x;
  if (idx >= M * 256) return;
  int c = idx % 256, m = idx / 256;
  xp[idx] = xz[(size_t)m * 512 + c];
  z[idx]  = xz[(size_t)m * 512 + 256 + c];
}

// LayerNorm over last dim; grid = rows, block = C (128 or 256)
__global__ void k_layernorm(const float* __restrict__ x, const float* __restrict__ g,
                            const float* __restrict__ b, float* __restrict__ y, int C) {
  __shared__ float red[256];
  const int row = blockIdx.x, c = threadIdx.x;
  const float v = x[(size_t)row * C + c];
  red[c] = v; __syncthreads();
  for (int s = blockDim.x >> 1; s > 0; s >>= 1) { if (c < s) red[c] += red[c + s]; __syncthreads(); }
  const float mu = red[0] * (1.f / C);
  __syncthreads();
  const float d = v - mu;
  red[c] = d * d; __syncthreads();
  for (int s = blockDim.x >> 1; s > 0; s >>= 1) { if (c < s) red[c] += red[c + s]; __syncthreads(); }
  const float var = red[0] * (1.f / C);
  y[(size_t)row * C + c] = d * rsqrtf(var + 1e-5f) * g[c] + b[c];
}

// Depthwise 3x3 NHWC, optional bias, optional fused BN, activation
__global__ void k_dwconv3x3(const float* __restrict__ x, const float* __restrict__ w,
                            const float* __restrict__ bias,
                            const float* __restrict__ bng, const float* __restrict__ bnb,
                            const float* __restrict__ bnm, const float* __restrict__ bnv,
                            float* __restrict__ y, int B, int H, int W, int C, int act) {
  int idx = blockIdx.x * blockDim.x + threadIdx.x;
  const int total = B * H * W * C;
  if (idx >= total) return;
  int c = idx % C; int t = idx / C; int wp = t % W; t /= W; int h = t % H; int b = t / H;
  float acc = bias ? bias[c] : 0.f;
#pragma unroll
  for (int ky = 0; ky < 3; ++ky)
#pragma unroll
    for (int kx = 0; kx < 3; ++kx) {
      const int hh = h + ky - 1, ww = wp + kx - 1;
      if (hh >= 0 && hh < H && ww >= 0 && ww < W)
        acc += x[((size_t)(b * H + hh) * W + ww) * C + c] * w[c * 9 + ky * 3 + kx];
    }
  if (bng) { const float s = bng[c] * rsqrtf(bnv[c] + 1e-5f); acc = acc * s + (bnb[c] - bnm[c] * s); }
  y[idx] = act_apply(acc, act);
}

// Build 4-direction scan inputs: xs laid out [k][B][L][256]
__global__ void k_prepare_xs(const float* __restrict__ xc, float* __restrict__ xs,
                             int B, int H, int W) {
  int idx = blockIdx.x * blockDim.x + threadIdx.x;
  const int total = B * H * W * 256;
  if (idx >= total) return;
  int d = idx & 255; int t = idx >> 8;
  int w = t % W; t /= W; int h = t % H; int b = t / H;
  const int L = H * W;
  const int l0 = h * W + w, l1 = w * H + h;
  const float v = xc[idx];
  xs[((size_t)(0 * B + b) * L + l0) * 256 + d] = v;
  xs[((size_t)(1 * B + b) * L + l1) * 256 + d] = v;
  xs[((size_t)(2 * B + b) * L + (L - 1 - l0)) * 256 + d] = v;
  xs[((size_t)(3 * B + b) * L + (L - 1 - l1)) * 256 + d] = v;
}

// Selective scan: block per (k,b), 256 threads = d channels, sequential over L.
// xdbl rows: [..., 0:8]=dts, [8:16]=Bs, [16:24]=Cs
__global__ void k_scan(const float* __restrict__ dt, const float* __restrict__ xs,
                       const float* __restrict__ xdbl,
                       const float* __restrict__ A_log, const float* __restrict__ Ds,
                       float* __restrict__ y, int B, int L) {
  const int kb = blockIdx.x;
  const int k = kb / B;
  const int d = threadIdx.x;
  float Av[8];
#pragma unroll
  for (int n = 0; n < 8; ++n) Av[n] = -__expf(A_log[((size_t)k * 256 + d) * 8 + n]);
  const float Dval = Ds[k * 256 + d];
  float h[8] = {0.f, 0.f, 0.f, 0.f, 0.f, 0.f, 0.f, 0.f};
  const size_t rowbase = (size_t)kb * L;
  for (int l = 0; l < L; ++l) {
    const size_t r = rowbase + l;
    const float dtv = dt[r * 256 + d];
    const float xv  = xs[r * 256 + d];
    const float* xd = xdbl + r * 24;
    const float dx = dtv * xv;
    float acc = 0.f;
#pragma unroll
    for (int n = 0; n < 8; ++n) {
      const float a = __expf(dtv * Av[n]);
      h[n] = a * h[n] + dx * xd[8 + n];
      acc += h[n] * xd[16 + n];
    }
    y[r * 256 + d] = acc + xv * Dval;
  }
}

// Merge 4 scan directions back into NHWC pixel-major [B*L, 256]
__global__ void k_merge(const float* __restrict__ ys, float* __restrict__ out,
                        int B, int H, int W) {
  int idx = blockIdx.x * blockDim.x + threadIdx.x;
  const int total = B * H * W * 256;
  if (idx >= total) return;
  int d = idx & 255; int t = idx >> 8;
  int w = t % W; t /= W; int h = t % H; int b = t / H;
  const int L = H * W;
  const int l0 = h * W + w, l1 = w * H + h;
  float v = ys[((size_t)(0 * B + b) * L + l0) * 256 + d]
          + ys[((size_t)(2 * B + b) * L + (L - 1 - l0)) * 256 + d]
          + ys[((size_t)(1 * B + b) * L + l1) * 256 + d]
          + ys[((size_t)(3 * B + b) * L + (L - 1 - l1)) * 256 + d];
  out[((size_t)(b * L + l0)) * 256 + d] = v;
}

// im2col for 3x3 pad=1: NHWC [B,H,W,C] -> [B*H*W, C*9], K index = c*9+ky*3+kx
__global__ void k_im2col3x3(const float* __restrict__ x, float* __restrict__ col,
                            int B, int H, int W, int C) {
  int idx = blockIdx.x * blockDim.x + threadIdx.x;
  const int total = B * H * W * C;
  if (idx >= total) return;
  int c = idx % C; int t = idx / C; int w = t % W; t /= W; int h = t % H; int b = t / H;
  const size_t m = (size_t)(b * H + h) * W + w;
  float* dst = col + m * (size_t)C * 9 + (size_t)c * 9;
#pragma unroll
  for (int ky = 0; ky < 3; ++ky)
#pragma unroll
    for (int kx = 0; kx < 3; ++kx) {
      const int hh = h + ky - 1, ww = w + kx - 1;
      float v = 0.f;
      if (hh >= 0 && hh < H && ww >= 0 && ww < W)
        v = x[((size_t)(b * H + hh) * W + ww) * C + c];
      dst[ky * 3 + kx] = v;
    }
}

// Channel attention: block per (b, head). qkv NHWC [B,L,384], hd=16, L<=16.
__global__ void k_attn(const float* __restrict__ qkv, const float* __restrict__ temp,
                       float* __restrict__ out, int B, int L) {
  const int bh = blockIdx.x;
  const int b = bh >> 3, hd = bh & 7;
  __shared__ float q[16][17], kk[16][17], v[16][17], at[16][17];
  const int tid = threadIdx.x;
  const int i = tid >> 4, l = tid & 15;
  float qv = 0.f, kv = 0.f, vv = 0.f;
  if (l < L) {
    const int cb = hd * 16 + i;
    const size_t base = ((size_t)b * L + l) * 384;
    qv = qkv[base + cb];
    kv = qkv[base + 128 + cb];
    vv = qkv[base + 256 + cb];
  }
  q[i][l] = qv; kk[i][l] = kv; v[i][l] = vv;
  __syncthreads();
  float qn = 0.f, kn = 0.f, dot = 0.f;
  for (int t2 = 0; t2 < L; ++t2) {
    qn += q[i][t2] * q[i][t2];
    kn += kk[l][t2] * kk[l][t2];
    dot += q[i][t2] * kk[l][t2];
  }
  qn = fmaxf(sqrtf(qn), 1e-12f);
  kn = fmaxf(sqrtf(kn), 1e-12f);
  at[i][l] = (dot / (qn * kn)) * temp[hd];
  __syncthreads();
  float mx = -1e30f;
  for (int j = 0; j < 16; ++j) mx = fmaxf(mx, at[i][j]);
  float se = 0.f;
  for (int j = 0; j < 16; ++j) se += __expf(at[i][j] - mx);
  const float wgt = __expf(at[i][l] - mx) / se;
  __syncthreads();
  at[i][l] = wgt;
  __syncthreads();
  if (l < L) {
    float o = 0.f;
    for (int j = 0; j < 16; ++j) o += at[i][j] * v[j][l];
    out[((size_t)b * L + l) * 128 + hd * 16 + i] = o;
  }
}

// global average pool: block per b, threads = C
__global__ void k_pool(const float* __restrict__ x, float* __restrict__ out, int L, int C) {
  const int b = blockIdx.x, c = threadIdx.x;
  float s = 0.f;
  for (int l = 0; l < L; ++l) s += x[((size_t)b * L + l) * C + c];
  out[b * C + c] = s / (float)L;
}

__global__ void k_se_apply(const float* __restrict__ y, const float* __restrict__ se,
                           const float* __restrict__ resid, float* __restrict__ out,
                           int B, int L, int C) {
  int idx = blockIdx.x * blockDim.x + threadIdx.x;
  const int total = B * L * C;
  if (idx >= total) return;
  int c = idx % C; int t = idx / C; int b = t / L;
  float v = y[idx] * se[b * C + c] + resid[idx];
  out[idx] = v > 0.f ? v : 0.f;
}

// bilinear 2x upsample (half-pixel centers) + add; NHWC; up:[B,h,w,C] base/out:[B,2h,2w,C]
__global__ void k_up_add(const float* __restrict__ up, const float* __restrict__ base,
                         float* __restrict__ out, int B, int h, int w, int C) {
  int idx = blockIdx.x * blockDim.x + threadIdx.x;
  const int H = 2 * h, W = 2 * w;
  const int total = B * H * W * C;
  if (idx >= total) return;
  int c = idx % C; int t = idx / C; int ow = t % W; t /= W; int oh = t % H; int b = t / H;
  float fy = (oh + 0.5f) * 0.5f - 0.5f;
  float fx = (ow + 0.5f) * 0.5f - 0.5f;
  fy = fminf(fmaxf(fy, 0.f), (float)(h - 1));
  fx = fminf(fmaxf(fx, 0.f), (float)(w - 1));
  int y0 = (int)floorf(fy), x0 = (int)floorf(fx);
  int y1 = min(y0 + 1, h - 1), x1 = min(x0 + 1, w - 1);
  const float wy = fy - (float)y0, wx = fx - (float)x0;
  const size_t rb = (size_t)b * h * w * C;
  const float v00 = up[rb + ((size_t)y0 * w + x0) * C + c];
  const float v01 = up[rb + ((size_t)y0 * w + x1) * C + c];
  const float v10 = up[rb + ((size_t)y1 * w + x0) * C + c];
  const float v11 = up[rb + ((size_t)y1 * w + x1) * C + c];
  const float v = v00 * (1 - wy) * (1 - wx) + v01 * (1 - wy) * wx
                + v10 * wy * (1 - wx) + v11 * wy * wx;
  out[idx] = v + base[idx];
}

// q = o2[...,0::2] + o2[...,1::2] + o3[...,:W] + o3[...,W:]
__global__ void k_stage_q(const float* __restrict__ o2, const float* __restrict__ o3,
                          float* __restrict__ q, int B, int H, int W) {
  int idx = blockIdx.x * blockDim.x + threadIdx.x;
  const int total = B * H * W * 128;
  if (idx >= total) return;
  int c = idx % 128; int t = idx / 128; int w = t % W; t /= W; int h = t % H; int b = t / H;
  const size_t row2 = (size_t)(b * H + h) * (2 * W);
  q[idx] = o2[(row2 + 2 * w) * 128 + c] + o2[(row2 + 2 * w + 1) * 128 + c]
         + o3[(row2 + w) * 128 + c] + o3[(row2 + w + W) * 128 + c];
}

__global__ void k_nhwc_to_nchw(const float* __restrict__ x, float* __restrict__ out,
                               int B, int H, int W, int C) {
  int idx = blockIdx.x * blockDim.x + threadIdx.x;
  const int total = B * H * W * C;
  if (idx >= total) return;
  int c = idx % C; int t = idx / C; int w = t % W; t /= W; int h = t % H; int b = t / H;
  out[((size_t)(b * C + c) * H + h) * W + w] = x[idx];
}

// ---------------------------------------------------------------------------
// Host side
// ---------------------------------------------------------------------------
#define F(p) ((const float*)(p))

struct Arena {
  char* base; size_t off, cap;
  float* alloc(size_t elems) {
    size_t bytes = (elems * sizeof(float) + 255) & ~(size_t)255;
    float* p = (float*)(base + off);
    off += bytes;
    return p;
  }
};

struct MambaP { const float *ln_g,*ln_b,*in_proj,*conv_w,*conv_b,*x_proj_w,*dt_proj_w,
                *dt_proj_b,*A_log,*Ds,*out_norm_g,*out_norm_b,*out_proj; };
struct StP    { const float *cw,*cb; MambaP m; };
struct AttnP  { const float *temperature,*qkv_w,*qkv_dw_w,*proj_w; };
struct StageP { StP st1, st2, st3; const float* dc_w; AttnP attn; bool has_attn; };
struct FuseP  { const float *w,*b,*bng,*bnb,*bnm,*bnv; };
struct IcsfP  { const float *conv1_w,*bn1g,*bn1b,*bn1m,*bn1v,*dw_w,*bndg,*bndb,*bndm,*bndv,
                *se1_w,*se1_b,*se2_w,*se2_b; };

static StP load_st(void* const* d, int& i) {
  StP s; s.cw = F(d[i++]); s.cb = F(d[i++]);
  MambaP& m = s.m;
  m.ln_g = F(d[i++]); m.ln_b = F(d[i++]); m.in_proj = F(d[i++]);
  m.conv_w = F(d[i++]); m.conv_b = F(d[i++]); m.x_proj_w = F(d[i++]);
  m.dt_proj_w = F(d[i++]); m.dt_proj_b = F(d[i++]); m.A_log = F(d[i++]); m.Ds = F(d[i++]);
  m.out_norm_g = F(d[i++]); m.out_norm_b = F(d[i++]); m.out_proj = F(d[i++]);
  return s;
}
static StageP load_stage(void* const* d, int& i, bool attn) {
  StageP sp; sp.st1 = load_st(d, i); sp.st2 = load_st(d, i); sp.st3 = load_st(d, i);
  sp.has_attn = attn; sp.dc_w = nullptr;
  sp.attn.temperature = sp.attn.qkv_w = sp.attn.qkv_dw_w = sp.attn.proj_w = nullptr;
  if (attn) {
    sp.dc_w = F(d[i++]);
    sp.attn.temperature = F(d[i++]); sp.attn.qkv_w = F(d[i++]);
    sp.attn.qkv_dw_w = F(d[i++]); sp.attn.proj_w = F(d[i++]);
  }
  return sp;
}
static FuseP load_fuse(void* const* d, int& i) {
  FuseP f; f.w = F(d[i++]); f.b = F(d[i++]);
  f.bng = F(d[i++]); f.bnb = F(d[i++]); f.bnm = F(d[i++]); f.bnv = F(d[i++]);
  return f;
}
static IcsfP load_icsf(void* const* d, int& i) {
  IcsfP p;
  p.conv1_w = F(d[i++]);
  p.bn1g = F(d[i++]); p.bn1b = F(d[i++]); p.bn1m = F(d[i++]); p.bn1v = F(d[i++]);
  p.dw_w = F(d[i++]);
  p.bndg = F(d[i++]); p.bndb = F(d[i++]); p.bndm = F(d[i++]); p.bndv = F(d[i++]);
  p.se1_w = F(d[i++]); p.se1_b = F(d[i++]); p.se2_w = F(d[i++]); p.se2_b = F(d[i++]);
  return p;
}

static inline unsigned gsz(long long total) { return (unsigned)((total + 255) / 256); }

static void gemm(hipStream_t s, const float* A, int lda, const float* Bm, int bT,
                 const float* bias, const float* res,
                 const float* bng, const float* bnb, const float* bnm, const float* bnv,
                 float* Cout, int M, int N, int K, int act) {
  const int tiles = ((M + 15) / 16) * ((N + 31) / 32);
  const int blocks = (tiles + 3) / 4;
  k_gemm<<<blocks, 128, 0, s>>>(A, lda, Bm, bT, bias, res, bng, bnb, bnm, bnv,
                                Cout, M, N, K, act);
}

static void dwconv(hipStream_t s, const float* x, const float* w, const float* bias,
                   const float* g, const float* b, const float* m, const float* v,
                   float* y, int B, int H, int W, int C, int act) {
  long long total = (long long)B * H * W * C;
  k_dwconv3x3<<<gsz(total), 256, 0, s>>>(x, w, bias, g, b, m, v, y, B, H, W, C, act);
}

// VSS (SS2D) block: xin NHWC [B*H*W, Cin] -> out NHWC [B*H*W, 128]
static void run_st_block(hipStream_t s, Arena& ar, const StP& p,
                         const float* xin, int B, int H, int W, int Cin, float* out) {
  const size_t ck = ar.off;
  const int M = B * H * W;      // also B*L
  const int L = H * W;

  float* y0 = ar.alloc((size_t)M * 128);
  gemm(s, xin, Cin, p.cw, 1, p.cb, nullptr, nullptr, nullptr, nullptr, nullptr,
       y0, M, 128, Cin, ACT_NONE);

  float* ln = ar.alloc((size_t)M * 128);
  k_layernorm<<<M, 128, 0, s>>>(y0, p.m.ln_g, p.m.ln_b, ln, 128);

  float* xz = ar.alloc((size_t)M * 512);
  gemm(s, ln, 128, p.m.in_proj, 0, nullptr, nullptr, nullptr, nullptr, nullptr, nullptr,
       xz, M, 512, 128, ACT_NONE);

  float* xp = ar.alloc((size_t)M * 256);
  float* z  = ar.alloc((size_t)M * 256);
  k_split<<<gsz((long long)M * 256), 256, 0, s>>>(xz, xp, z, M);

  float* xc = ar.alloc((size_t)M * 256);
  dwconv(s, xp, p.m.conv_w, p.m.conv_b, nullptr, nullptr, nullptr, nullptr,
         xc, B, H, W, 256, ACT_SILU);

  float* xs = ar.alloc((size_t)4 * M * 256);
  k_prepare_xs<<<gsz((long long)M * 256), 256, 0, s>>>(xc, xs, B, H, W);

  float* xdbl = ar.alloc((size_t)4 * M * 24);
  for (int k = 0; k < 4; ++k)
    gemm(s, xs + (size_t)k * M * 256, 256, p.m.x_proj_w + (size_t)k * 256 * 24, 0,
         nullptr, nullptr, nullptr, nullptr, nullptr, nullptr,
         xdbl + (size_t)k * M * 24, M, 24, 256, ACT_NONE);

  float* dt = ar.alloc((size_t)4 * M * 256);
  for (int k = 0; k < 4; ++k)
    gemm(s, xdbl + (size_t)k * M * 24, 24, p.m.dt_proj_w + (size_t)k * 8 * 256, 0,
         p.m.dt_proj_b + (size_t)k * 256, nullptr, nullptr, nullptr, nullptr, nullptr,
         dt + (size_t)k * M * 256, M, 256, 8, ACT_SOFTPLUS);

  float* ys = ar.alloc((size_t)4 * M * 256);
  k_scan<<<4 * B, 256, 0, s>>>(dt, xs, xdbl, p.m.A_log, p.m.Ds, ys, B, L);

  float* mg = ar.alloc((size_t)M * 256);
  k_merge<<<gsz((long long)M * 256), 256, 0, s>>>(ys, mg, B, H, W);

  float* ln2 = ar.alloc((size_t)M * 256);
  k_layernorm<<<M, 256, 0, s>>>(mg, p.m.out_norm_g, p.m.out_norm_b, ln2, 256);

  float* gt = ar.alloc((size_t)M * 256);
  k_mul_silu<<<gsz((long long)M * 256), 256, 0, s>>>(ln2, z, gt, M * 256);

  gemm(s, gt, 256, p.m.out_proj, 0, nullptr, /*res=*/y0,
       nullptr, nullptr, nullptr, nullptr, out, M, 128, 256, ACT_NONE);

  ar.off = ck;
}

static void run_stage(hipStream_t s, Arena& ar, const StageP& sp,
                      const float* preN, const float* postN,
                      int B, int C, int H, int W, float* p1, float* q) {
  const size_t ck0 = ar.off;
  const int M = B * H * W;

  float* catN = ar.alloc((size_t)M * 2 * C);
  k_cat_c_nhwc<<<gsz((long long)M * 2 * C), 256, 0, s>>>(preN, postN, catN, B, C, H, W);

  run_st_block(s, ar, sp.st1, catN, B, H, W, 2 * C, p1);

  if (sp.has_attn) {
    const size_t ck = ar.off;
    const int Ka = 2 * C * 9;
    float* col = ar.alloc((size_t)M * Ka);
    k_im2col3x3<<<gsz((long long)M * 2 * C), 256, 0, s>>>(catN, col, B, H, W, 2 * C);
    float* a = ar.alloc((size_t)M * 128);
    gemm(s, col, Ka, sp.dc_w, 1, nullptr, nullptr, nullptr, nullptr, nullptr, nullptr,
         a, M, 128, Ka, ACT_NONE);
    float* qkv = ar.alloc((size_t)M * 384);
    gemm(s, a, 128, sp.attn.qkv_w, 1, nullptr, nullptr, nullptr, nullptr, nullptr, nullptr,
         qkv, M, 384, 128, ACT_NONE);
    float* qkvd = ar.alloc((size_t)M * 384);
    dwconv(s, qkv, sp.attn.qkv_dw_w, nullptr, nullptr, nullptr, nullptr, nullptr,
           qkvd, B, H, W, 384, ACT_NONE);
    float* ao = ar.alloc((size_t)M * 128);
    k_attn<<<B * 8, 256, 0, s>>>(qkvd, sp.attn.temperature, ao, B, H * W);
    float* a2 = ar.alloc((size_t)M * 128);
    gemm(s, ao, 128, sp.attn.proj_w, 1, nullptr, /*res=*/a,
         nullptr, nullptr, nullptr, nullptr, a2, M, 128, 128, ACT_NONE);
    k_add<<<gsz((long long)M * 128), 256, 0, s>>>(p1, a2, p1, M * 128);
    ar.off = ck;
  }

  // q path: ct2 (interleaved W) and ct3 (concat W), both NHWC [B,H,2W,C]
  float* ct2 = ar.alloc((size_t)M * 2 * C);
  k_stackW_nhwc<<<gsz((long long)M * 2 * C), 256, 0, s>>>(preN, postN, ct2, B, C, H, W);
  float* o2 = ar.alloc((size_t)2 * M * 128);
  run_st_block(s, ar, sp.st2, ct2, B, H, 2 * W, C, o2);

  float* ct3 = ar.alloc((size_t)M * 2 * C);
  k_catW_nhwc<<<gsz((long long)M * 2 * C), 256, 0, s>>>(preN, postN, ct3, B, C, H, W);
  float* o3 = ar.alloc((size_t)2 * M * 128);
  run_st_block(s, ar, sp.st3, ct3, B, H, 2 * W, C, o3);

  k_stage_q<<<gsz((long long)M * 128), 256, 0, s>>>(o2, o3, q, B, H, W);
  ar.off = ck0;
}

static void run_icsf(hipStream_t s, Arena& ar, const IcsfP& p,
                     const float* x, int B, int H, int W, float* out) {
  const size_t ck = ar.off;
  const int M = B * H * W, L = H * W;
  float* col = ar.alloc((size_t)M * 1152);
  k_im2col3x3<<<gsz((long long)M * 128), 256, 0, s>>>(x, col, B, H, W, 128);
  float* o1 = ar.alloc((size_t)M * 128);
  gemm(s, col, 1152, p.conv1_w, 1, nullptr, nullptr, p.bn1g, p.bn1b, p.bn1m, p.bn1v,
       o1, M, 128, 1152, ACT_RELU);
  float* o2 = ar.alloc((size_t)M * 128);
  dwconv(s, o1, p.dw_w, nullptr, p.bndg, p.bndb, p.bndm, p.bndv, o2, B, H, W, 128, ACT_RELU);
  float* pool = ar.alloc((size_t)B * 128);
  k_pool<<<B, 128, 0, s>>>(o2, pool, L, 128);
  float* se1 = ar.alloc((size_t)B * 16);
  gemm(s, pool, 128, p.se1_w, 1, p.se1_b, nullptr, nullptr, nullptr, nullptr, nullptr,
       se1, B, 16, 128, ACT_RELU);
  float* se2 = ar.alloc((size_t)B * 128);
  gemm(s, se1, 16, p.se2_w, 1, p.se2_b, nullptr, nullptr, nullptr, nullptr, nullptr,
       se2, B, 128, 16, ACT_SIGMOID);
  k_se_apply<<<gsz((long long)M * 128), 256, 0, s>>>(o2, se2, x, out, B, L, 128);
  ar.off = ck;
}

extern "C" void kernel_launch(void* const* d_in, const int* in_sizes, int n_in,
                              void* d_out, int out_size, void* d_ws, size_t ws_size,
                              hipStream_t stream) {
  (void)in_sizes; (void)n_in; (void)out_size;
  // Inputs 0..7: pre_feat_1, post_feat_1, ..., pre_feat_4, post_feat_4 (NCHW f32).
  // Params flattened depth-first in setup_inputs() insertion order, starting at 8.
  const float* pre1  = F(d_in[0]); const float* post1 = F(d_in[1]);
  const float* pre2  = F(d_in[2]); const float* post2 = F(d_in[3]);
  const float* pre3  = F(d_in[4]); const float* post3 = F(d_in[5]);
  const float* pre4  = F(d_in[6]); const float* post4 = F(d_in[7]);

  int i = 8;
  StageP s4 = load_stage(d_in, i, true);
  StageP s3 = load_stage(d_in, i, true);
  StageP s2 = load_stage(d_in, i, false);
  StageP s1 = load_stage(d_in, i, false);
  FuseP f4 = load_fuse(d_in, i), f3 = load_fuse(d_in, i);
  FuseP f2 = load_fuse(d_in, i), f1 = load_fuse(d_in, i);
  IcsfP m3 = load_icsf(d_in, i), m2 = load_icsf(d_in, i), m1 = load_icsf(d_in, i);

  Arena ar{(char*)d_ws, 0, ws_size};
  hipStream_t s = stream;
  const int B = 4;
  const int M4 = B * 2 * 2, M3 = B * 4 * 4, M2 = B * 8 * 8, M1 = B * 16 * 16;

  // ---- stage 4 (C=768, 2x2, attn) ----
  float* p41 = ar.alloc((size_t)M4 * 128);
  float* q4  = ar.alloc((size_t)M4 * 128);
  run_stage(s, ar, s4, pre4, post4, B, 768, 2, 2, p41, q4);
  float* cat4 = ar.alloc((size_t)M4 * 256);
  k_cat2<<<gsz((long long)M4 * 256), 256, 0, s>>>(p41, q4, cat4, M4);
  float* p4 = ar.alloc((size_t)M4 * 128);
  gemm(s, cat4, 256, f4.w, 1, f4.b, nullptr, f4.bng, f4.bnb, f4.bnm, f4.bnv,
       p4, M4, 128, 256, ACT_RELU);

  // ---- stage 3 (C=384, 4x4, attn) ----
  float* p31 = ar.alloc((size_t)M3 * 128);
  float* q3  = ar.alloc((size_t)M3 * 128);
  run_stage(s, ar, s3, pre3, post3, B, 384, 4, 4, p31, q3);
  float* cat3 = ar.alloc((size_t)M3 * 256);
  k_cat2<<<gsz((long long)M3 * 256), 256, 0, s>>>(p31, q3, cat3, M3);
  float* p3 = ar.alloc((size_t)M3 * 128);
  gemm(s, cat3, 256, f3.w, 1, f3.b, nullptr, f3.bng, f3.bnb, f3.bnm, f3.bnv,
       p3, M3, 128, 256, ACT_RELU);

  // ---- stage 2 (C=192, 8x8) ----
  float* p21 = ar.alloc((size_t)M2 * 128);
  float* q2  = ar.alloc((size_t)M2 * 128);
  run_stage(s, ar, s2, pre2, post2, B, 192, 8, 8, p21, q2);
  float* sum2 = ar.alloc((size_t)M2 * 128);
  k_add<<<gsz((long long)M2 * 128), 256, 0, s>>>(p21, q2, sum2, M2 * 128);
  float* p2 = ar.alloc((size_t)M2 * 128);
  gemm(s, sum2, 128, f2.w, 1, f2.b, nullptr, f2.bng, f2.bnb, f2.bnm, f2.bnv,
       p2, M2, 128, 128, ACT_RELU);

  // ---- stage 1 (C=96, 16x16) ----
  float* p11 = ar.alloc((size_t)M1 * 128);
  float* q1  = ar.alloc((size_t)M1 * 128);
  run_stage(s, ar, s1, pre1, post1, B, 96, 16, 16, p11, q1);
  float* sum1 = ar.alloc((size_t)M1 * 128);
  k_add<<<gsz((long long)M1 * 128), 256, 0, s>>>(p11, q1, sum1, M1 * 128);
  float* p1 = ar.alloc((size_t)M1 * 128);
  gemm(s, sum1, 128, f1.w, 1, f1.b, nullptr, f1.bng, f1.bnb, f1.bnm, f1.bnv,
       p1, M1, 128, 128, ACT_RELU);

  // ---- top-down decoder ----
  float* u3 = ar.alloc((size_t)M3 * 128);
  k_up_add<<<gsz((long long)M3 * 128), 256, 0, s>>>(p4, p3, u3, B, 2, 2, 128);
  float* p3n = ar.alloc((size_t)M3 * 128);
  run_icsf(s, ar, m3, u3, B, 4, 4, p3n);

  float* u2 = ar.alloc((size_t)M2 * 128);
  k_up_add<<<gsz((long long)M2 * 128), 256, 0, s>>>(p3n, p2, u2, B, 4, 4, 128);
  float* p2n = ar.alloc((size_t)M2 * 128);
  run_icsf(s, ar, m2, u2, B, 8, 8, p2n);

  float* u1 = ar.alloc((size_t)M1 * 128);
  k_up_add<<<gsz((long long)M1 * 128), 256, 0, s>>>(p2n, p1, u1, B, 8, 8, 128);
  float* p1n = ar.alloc((size_t)M1 * 128);
  run_icsf(s, ar, m1, u1, B, 16, 16, p1n);

  // final NHWC -> NCHW into d_out  [4,128,16,16]
  k_nhwc_to_nchw<<<gsz((long long)M1 * 128), 256, 0, s>>>(p1n, (float*)d_out, B, 16, 16, 128);
}
